// CrossScaleOrthogonalMambaBlock_84911503442587
// MI455X (gfx1250) — matrix operand order repaired and checked by reading, not compiled, not run on hardware
//
#include <hip/hip_runtime.h>
#include <hip/hip_bf16.h>
#include <math.h>

// Problem constants (match reference)
#define Bq   8
#define Lq   2048
#define Dq   512
#define Iq   256
#define Sq   8
#define DTRq 32
#define PHq  128
#define Mq   (Bq * Lq)          // 16384 rows

typedef __bf16 bf16_t;
typedef bf16_t v16bf __attribute__((ext_vector_type(16)));
typedef float  v8f   __attribute__((ext_vector_type(8)));

static __device__ __forceinline__ float softplusf(float x) {
    return (x > 20.f) ? x : log1pf(expf(x));
}
static __device__ __forceinline__ float siluf(float x) {
    return x / (1.f + expf(-x));
}

// ---------------------------------------------------------------------------
// K1: si = LayerNorm(x) + posMLP(position), written into cat[:, 1024:1536]
// one block (128 threads) per row
// ---------------------------------------------------------------------------
__global__ void __launch_bounds__(PHq)
si_kernel(const float* __restrict__ x, const float* __restrict__ pos,
          const float* __restrict__ lnw, const float* __restrict__ lnb,
          const float* __restrict__ w1, const float* __restrict__ b1,
          const float* __restrict__ w2, const float* __restrict__ b2,
          float* __restrict__ cat)
{
    const int row = blockIdx.x;
    const int tid = threadIdx.x;
    __shared__ float h[PHq];
    __shared__ float red[PHq];
    __shared__ float pv[6];
    if (tid < 6) pv[tid] = pos[(size_t)row * 6 + tid];
    __syncthreads();

    // hidden = gelu(position @ w1.T + b1), exact gelu
    float acc = b1[tid];
    #pragma unroll
    for (int j = 0; j < 6; ++j) acc += pv[j] * w1[tid * 6 + j];
    h[tid] = 0.5f * acc * (1.0f + erff(acc * 0.70710678118654752f));

    // LayerNorm stats over x row (512)
    float xv[4];
    float s = 0.f;
    #pragma unroll
    for (int q = 0; q < 4; ++q) {
        xv[q] = x[(size_t)row * Dq + tid + PHq * q];
        s += xv[q];
    }
    red[tid] = s; __syncthreads();
    for (int off = PHq / 2; off > 0; off >>= 1) {
        if (tid < off) red[tid] += red[tid + off];
        __syncthreads();
    }
    const float mean = red[0] * (1.f / Dq);
    __syncthreads();
    s = 0.f;
    #pragma unroll
    for (int q = 0; q < 4; ++q) { float d = xv[q] - mean; s += d * d; }
    red[tid] = s; __syncthreads();
    for (int off = PHq / 2; off > 0; off >>= 1) {
        if (tid < off) red[tid] += red[tid + off];
        __syncthreads();
    }
    const float rstd = rsqrtf(red[0] * (1.f / Dq) + 1e-6f);

    #pragma unroll
    for (int q = 0; q < 4; ++q) {
        const int dcol = tid + PHq * q;
        float p = b2[dcol];
        for (int ph = 0; ph < PHq; ++ph) p += h[ph] * w2[dcol * PHq + ph];
        const float v = (xv[q] - mean) * rstd * lnw[dcol] + lnb[dcol] + p;
        cat[(size_t)row * (3 * Dq) + 2 * Dq + dcol] = v;
    }
}

// ---------------------------------------------------------------------------
// Pack f32 matrix X [R x K] (row stride ldx) into A-side WMMA bf16 fragments.
// Fragment (tm, kt) = 16x32 tile; each lane's 16 elements stored contiguously:
//   out[((tm*KT + kt)*32 + lane)*16 + e]
// A layout (ISA 7.12.2): VGPR j<4 -> K = 2j + 8*hf; j>=4 -> K = 16+2(j-4)+8*hf
// One thread per (fragment, lane).
// ---------------------------------------------------------------------------
__global__ void __launch_bounds__(256)
pack_a_kernel(const float* __restrict__ X, int ldx,
              bf16_t* __restrict__ out, int KT)
{
    const int gid  = blockIdx.x * 256 + threadIdx.x;
    const int lane = gid & 31;
    const int frag = gid >> 5;
    const int kt = frag % KT;
    const int tm = frag / KT;
    const int ml = lane & 15, hf = lane >> 4;
    const float* __restrict__ p = X + (size_t)(tm * 16 + ml) * ldx + kt * 32;
    v16bf f;
    #pragma unroll
    for (int j = 0; j < 8; ++j) {
        const int ka = (j < 4) ? (2 * j + 8 * hf) : (16 + 2 * (j - 4) + 8 * hf);
        const float2 v = *(const float2*)(p + ka);
        f[2 * j]     = (bf16_t)v.x;
        f[2 * j + 1] = (bf16_t)v.y;
    }
    *(v16bf*)(out + (size_t)gid * 16) = f;
}

// ---------------------------------------------------------------------------
// Pack f32 weight W [N x K] (x @ W.T) into B-side WMMA bf16 fragments.
// B layout: lane's column n = lane%16 (= W row), j<4 -> K = 2j + 8*hf; j>=4 -> K = 16+2(j-4)+8*hf {,+1}.
// ---------------------------------------------------------------------------
__global__ void __launch_bounds__(256)
pack_w_kernel(const float* __restrict__ W, int ldw,
              bf16_t* __restrict__ out, int KT)
{
    const int gid  = blockIdx.x * 256 + threadIdx.x;
    const int lane = gid & 31;
    const int frag = gid >> 5;
    const int kt = frag % KT;
    const int tn = frag / KT;
    const int ml = lane & 15, hf = lane >> 4;
    const float* __restrict__ p = W + (size_t)(tn * 16 + ml) * ldw + kt * 32;
    v16bf f;
    #pragma unroll
    for (int j = 0; j < 8; ++j) {
        const int kb = (j < 4) ? (2 * j + 8 * hf) : (16 + 2 * (j - 4) + 8 * hf);
        const float2 v = *(const float2*)(p + kb);
        f[2 * j]     = (bf16_t)v.x;
        f[2 * j + 1] = (bf16_t)v.y;
    }
    *(v16bf*)(out + (size_t)gid * 16) = f;
}

// ---------------------------------------------------------------------------
// K2: GEMM on pre-packed bf16 fragments. C[M,N] = A @ W^T, f32 accumulate.
// Each wave computes a 16x64 strip: one A fragment feeds 4 WMMAs per k-step.
// Inner loop = pure global_load_b128 + v_wmma (no conversions).
// ---------------------------------------------------------------------------
__global__ void __launch_bounds__(256)
gemm_wmma_packed(const bf16_t* __restrict__ Apack,
                 const bf16_t* __restrict__ Wpack,
                 float* __restrict__ C, int ldc, int KT, int tilesN)
{
    const int lane = threadIdx.x & 31;
    const int wave = blockIdx.x * (blockDim.x >> 5) + (threadIdx.x >> 5);
    const int groupsN = tilesN >> 2;
    const int tm  = wave / groupsN;
    const int tn0 = (wave % groupsN) * 4;

    const v16bf* __restrict__ Av = (const v16bf*)Apack;
    const v16bf* __restrict__ Wv = (const v16bf*)Wpack;
    const v16bf* __restrict__ Ap = Av + (size_t)tm * KT * 32 + lane;
    const v16bf* __restrict__ W0 = Wv + (size_t)(tn0 + 0) * KT * 32 + lane;
    const v16bf* __restrict__ W1 = Wv + (size_t)(tn0 + 1) * KT * 32 + lane;
    const v16bf* __restrict__ W2 = Wv + (size_t)(tn0 + 2) * KT * 32 + lane;
    const v16bf* __restrict__ W3 = Wv + (size_t)(tn0 + 3) * KT * 32 + lane;

    v8f acc0 = {0.f,0.f,0.f,0.f,0.f,0.f,0.f,0.f};
    v8f acc1 = acc0, acc2 = acc0, acc3 = acc0;

    for (int kt = 0; kt < KT; ++kt) {
        if (kt + 2 < KT)
            __builtin_prefetch((const void*)(Ap + (size_t)(kt + 2) * 32), 0, 1);
        const v16bf af = Ap[(size_t)kt * 32];
        const v16bf w0 = W0[(size_t)kt * 32];
        const v16bf w1 = W1[(size_t)kt * 32];
        const v16bf w2 = W2[(size_t)kt * 32];
        const v16bf w3 = W3[(size_t)kt * 32];
        acc0 = __builtin_amdgcn_wmma_f32_16x16x32_bf16(false, af, false, w0, (short)0, acc0, false, false);
        acc1 = __builtin_amdgcn_wmma_f32_16x16x32_bf16(false, af, false, w1, (short)0, acc1, false, false);
        acc2 = __builtin_amdgcn_wmma_f32_16x16x32_bf16(false, af, false, w2, (short)0, acc2, false, false);
        acc3 = __builtin_amdgcn_wmma_f32_16x16x32_bf16(false, af, false, w3, (short)0, acc3, false, false);
    }

    // C tile layout: VGPR r -> row tm*16 + r + 8*hf, column = tile*16 + lane%16
    const int ml = lane & 15, hf = lane >> 4;
    #pragma unroll
    for (int r = 0; r < 8; ++r) {
        const size_t rowoff = (size_t)(tm * 16 + r + 8 * hf) * ldc;
        C[rowoff + (tn0 + 0) * 16 + ml] = acc0[r];
        C[rowoff + (tn0 + 1) * 16 + ml] = acc1[r];
        C[rowoff + (tn0 + 2) * 16 + ml] = acc2[r];
        C[rowoff + (tn0 + 3) * 16 + ml] = acc3[r];
    }
}

// ---------------------------------------------------------------------------
// K3: depthwise conv1d (K=3, same-pad) + SiLU for both xb and zb halves of xz.
// dir==1 uses reversed taps (conv of flipped sequence, algebraically).
// xbc gets silu(conv(xb)); zb result goes straight into yz[:, 256:512].
// ---------------------------------------------------------------------------
__global__ void __launch_bounds__(256)
conv_silu_kernel(const float* __restrict__ xz,
                 const float* __restrict__ cxw, const float* __restrict__ czw,
                 float* __restrict__ xbc, float* __restrict__ yz, int dir)
{
    const int idx = blockIdx.x * 256 + threadIdx.x;
    const int i   = idx & (Iq - 1);
    const int row = idx >> 8;
    const int l   = row & (Lq - 1);

    const int t0 = dir ? 2 : 0, t2 = dir ? 0 : 2;
    const float wx0 = cxw[i * 3 + t0], wx1 = cxw[i * 3 + 1], wx2 = cxw[i * 3 + t2];
    const float wz0 = czw[i * 3 + t0], wz1 = czw[i * 3 + 1], wz2 = czw[i * 3 + t2];

    const size_t base = (size_t)row * Dq;
    const float xm = (l > 0)      ? xz[base - Dq + i] : 0.f;
    const float xc = xz[base + i];
    const float xp = (l < Lq - 1) ? xz[base + Dq + i] : 0.f;
    const float zm = (l > 0)      ? xz[base - Dq + Iq + i] : 0.f;
    const float zc = xz[base + Iq + i];
    const float zp = (l < Lq - 1) ? xz[base + Dq + Iq + i] : 0.f;

    const float sx = xm * wx0 + xc * wx1 + xp * wx2;
    const float sz = zm * wz0 + zc * wz1 + zp * wz2;
    xbc[(size_t)row * Iq + i]     = siluf(sx);
    yz[(size_t)row * Dq + Iq + i] = siluf(sz);
}

// ---------------------------------------------------------------------------
// K4: proj = xbc @ xp_w.T (48 outs), delta = clip(softplus(proj[:32]@dt_w.T+b)),
// bt/ct = tanh of proj tails. One block (256 threads) per row.
// ---------------------------------------------------------------------------
__global__ void __launch_bounds__(256)
proj_delta_kernel(const float* __restrict__ xbc,
                  const float* __restrict__ xpw,   // [48,256]
                  const float* __restrict__ dtw,   // [256,32]
                  const float* __restrict__ dtb,   // [256]
                  float* __restrict__ delta, float* __restrict__ btb,
                  float* __restrict__ ctb)
{
    const int row = blockIdx.x;
    const int tid = threadIdx.x;
    __shared__ float xl[Iq];
    __shared__ float pr[DTRq + 2 * Sq];

    xl[tid] = xbc[(size_t)row * Iq + tid];
    __syncthreads();
    if (tid < DTRq + 2 * Sq) {
        float s = 0.f;
        for (int k = 0; k < Iq; ++k) s += xl[k] * xpw[tid * Iq + k];
        pr[tid] = s;
    }
    __syncthreads();

    float acc = dtb[tid];
    #pragma unroll
    for (int r = 0; r < DTRq; ++r) acc += pr[r] * dtw[tid * DTRq + r];
    float d = softplusf(acc);
    d = fminf(fmaxf(d, 1e-4f), 1.0f);
    delta[(size_t)row * Iq + tid] = d;

    if (tid < Sq) {
        btb[(size_t)row * Sq + tid] = tanhf(pr[DTRq + tid]);
        ctb[(size_t)row * Sq + tid] = tanhf(pr[DTRq + Sq + tid]);
    }
}

// ---------------------------------------------------------------------------
// K5: sequential selective scan. 1 block per batch, thread i owns channel i,
// 8 recurrent states live in registers. dir==1 iterates l backwards.
// y written into yz[:, 0:256]. Prefetch hides the next step's loads.
// ---------------------------------------------------------------------------
__global__ void __launch_bounds__(256)
scan_kernel(const float* __restrict__ delta, const float* __restrict__ xbc,
            const float* __restrict__ btb, const float* __restrict__ ctb,
            const float* __restrict__ Alog, const float* __restrict__ Dp,
            float* __restrict__ yz, int dir)
{
    const int b = blockIdx.x;
    const int i = threadIdx.x;

    float a[Sq], st[Sq];
    #pragma unroll
    for (int s = 0; s < Sq; ++s) {
        a[s]  = softplusf(Alog[i * Sq + s]) + 1e-4f;
        st[s] = 0.f;
    }
    const float dpv = Dp[i];

    for (int t = 0; t < Lq; ++t) {
        const int l = dir ? (Lq - 1 - t) : t;
        const size_t row = (size_t)b * Lq + l;
        if (t + 1 < Lq) {
            const int ln = dir ? (Lq - 2 - t) : (t + 1);
            const size_t rn = (size_t)b * Lq + ln;
            __builtin_prefetch(&delta[rn * Iq + i], 0, 0);
            __builtin_prefetch(&xbc[rn * Iq + i], 0, 0);
        }
        const float dv = delta[row * Iq + i];
        const float xv = xbc[row * Iq + i];
        float y = dpv * xv;
        #pragma unroll
        for (int s = 0; s < Sq; ++s) {
            const float btv = btb[row * Sq + s];
            const float ctv = ctb[row * Sq + s];
            float dec = expf(-dv * a[s]);
            dec = fminf(fmaxf(dec, 1e-4f), 1.0f);
            st[s] = dec * st[s] + (1.f - dec) * (btv * xv);
            y += st[s] * ctv;
        }
        yz[row * Dq + i] = y;
    }
}

// ---------------------------------------------------------------------------
// K6: output LayerNorm. One block (256 threads) per row of mixed[M,512].
// ---------------------------------------------------------------------------
__global__ void __launch_bounds__(256)
ln_out_kernel(const float* __restrict__ mixed, const float* __restrict__ w,
              const float* __restrict__ b, float* __restrict__ out)
{
    const int row = blockIdx.x;
    const int tid = threadIdx.x;
    __shared__ float red[256];
    const float v0 = mixed[(size_t)row * Dq + tid];
    const float v1 = mixed[(size_t)row * Dq + 256 + tid];
    red[tid] = v0 + v1; __syncthreads();
    for (int off = 128; off > 0; off >>= 1) {
        if (tid < off) red[tid] += red[tid + off];
        __syncthreads();
    }
    const float mean = red[0] * (1.f / Dq);
    __syncthreads();
    const float d0 = v0 - mean, d1 = v1 - mean;
    red[tid] = d0 * d0 + d1 * d1; __syncthreads();
    for (int off = 128; off > 0; off >>= 1) {
        if (tid < off) red[tid] += red[tid + off];
        __syncthreads();
    }
    const float rstd = rsqrtf(red[0] * (1.f / Dq) + 1e-6f);
    out[(size_t)row * Dq + tid]       = d0 * rstd * w[tid] + b[tid];
    out[(size_t)row * Dq + 256 + tid] = d1 * rstd * w[256 + tid] + b[256 + tid];
}

// ---------------------------------------------------------------------------
extern "C" void kernel_launch(void* const* d_in, const int* in_sizes, int n_in,
                              void* d_out, int out_size, void* d_ws, size_t ws_size,
                              hipStream_t stream) {
    (void)in_sizes; (void)n_in; (void)out_size; (void)ws_size;
    const float* x        = (const float*)d_in[0];
    const float* position = (const float*)d_in[1];
    const float* ln_in_w  = (const float*)d_in[2];
    const float* ln_in_b  = (const float*)d_in[3];
    const float* pos_w1   = (const float*)d_in[4];
    const float* pos_b1   = (const float*)d_in[5];
    const float* pos_w2   = (const float*)d_in[6];
    const float* pos_b2   = (const float*)d_in[7];
    const float* in_w     = (const float*)d_in[8];    // [2,512,512]
    const float* cx_w     = (const float*)d_in[9];    // [2,256,1,3]
    const float* cz_w     = (const float*)d_in[10];   // [2,256,1,3]
    const float* xp_w     = (const float*)d_in[11];   // [2,48,256]
    const float* dt_w     = (const float*)d_in[12];   // [2,256,32]
    const float* dt_b     = (const float*)d_in[13];   // [2,256]
    const float* Alog     = (const float*)d_in[14];   // [2,256,8]
    const float* Dp       = (const float*)d_in[15];   // [2,256]
    const float* out_w    = (const float*)d_in[16];   // [2,512,512]
    const float* mix_w    = (const float*)d_in[17];   // [512,1536]
    const float* ln_out_w = (const float*)d_in[18];
    const float* ln_out_b = (const float*)d_in[19];
    float* outp = (float*)d_out;

    // workspace carve-up
    char* ws = (char*)d_ws;
    float* cat   = (float*)ws;  ws += (size_t)Mq * 1536 * 4;  // [fwd|bwd|si] f32
    float* xz    = (float*)ws;  ws += (size_t)Mq * Dq  * 4;   // reused as 'mixed'
    float* xbc   = (float*)ws;  ws += (size_t)Mq * Iq  * 4;
    float* yz    = (float*)ws;  ws += (size_t)Mq * Dq  * 4;   // [y|zb] per direction
    float* delta = (float*)ws;  ws += (size_t)Mq * Iq  * 4;
    float* btb   = (float*)ws;  ws += (size_t)Mq * Sq  * 4;
    float* ctb   = (float*)ws;  ws += (size_t)Mq * Sq  * 4;
    bf16_t* sipack  = (bf16_t*)ws; ws += (size_t)Mq * Dq * 2;        // si A-fragments
    bf16_t* yzpack  = (bf16_t*)ws; ws += (size_t)Mq * Dq * 2;        // yz A-fragments
    bf16_t* catpack = (bf16_t*)ws; ws += (size_t)Mq * 3 * Dq * 2;    // cat A-fragments
    bf16_t* inwp0   = (bf16_t*)ws; ws += (size_t)Dq * Dq * 2;
    bf16_t* inwp1   = (bf16_t*)ws; ws += (size_t)Dq * Dq * 2;
    bf16_t* outwp0  = (bf16_t*)ws; ws += (size_t)Dq * Dq * 2;
    bf16_t* outwp1  = (bf16_t*)ws; ws += (size_t)Dq * Dq * 2;
    bf16_t* mixwp   = (bf16_t*)ws; ws += (size_t)Dq * 3 * Dq * 2;
    bf16_t* inwp[2]  = {inwp0, inwp1};
    bf16_t* outwp[2] = {outwp0, outwp1};
    float* mixed = xz;  // xz no longer needed once both directions are done

    const int KT512  = Dq / 32;        // 16
    const int KT1536 = 3 * Dq / 32;    // 48
    const int tilesN = Dq / 16;        // 32
    const int gemmBlks = ((Mq / 16) * (tilesN / 4)) / 8;           // 1024
    const int packA512Blks  = ((Mq / 16) * KT512  * 32) / 256;     // 2048
    const int packA1536Blks = ((Mq / 16) * KT1536 * 32) / 256;     // 6144
    const int packW512Blks  = ((Dq / 16) * KT512  * 32) / 256;     // 64
    const int packW1536Blks = ((Dq / 16) * KT1536 * 32) / 256;     // 192

    // 0) pack all weights to bf16 B-fragments (tiny, L2-resident)
    pack_w_kernel<<<packW512Blks, 256, 0, stream>>>(in_w,               Dq, inwp0,  KT512);
    pack_w_kernel<<<packW512Blks, 256, 0, stream>>>(in_w + Dq * Dq,     Dq, inwp1,  KT512);
    pack_w_kernel<<<packW512Blks, 256, 0, stream>>>(out_w,              Dq, outwp0, KT512);
    pack_w_kernel<<<packW512Blks, 256, 0, stream>>>(out_w + Dq * Dq,    Dq, outwp1, KT512);
    pack_w_kernel<<<packW1536Blks, 256, 0, stream>>>(mix_w,          3 * Dq, mixwp,  KT1536);

    // 1) si = LN(x) + posMLP -> cat[:, 1024:1536], then pack once (shared by
    //    both directions' in-projections)
    si_kernel<<<Mq, PHq, 0, stream>>>(x, position, ln_in_w, ln_in_b,
                                      pos_w1, pos_b1, pos_w2, pos_b2, cat);
    pack_a_kernel<<<packA512Blks, 256, 0, stream>>>(cat + 2 * Dq, 3 * Dq, sipack, KT512);

    for (int d = 0; d < 2; ++d) {
        // 2a) in-projection: xz = si @ in_w[d].T
        gemm_wmma_packed<<<gemmBlks, 256, 0, stream>>>(
            sipack, inwp[d], xz, Dq, KT512, tilesN);
        // 2b) depthwise conv + silu (reversed taps for backward direction)
        conv_silu_kernel<<<(Mq * Iq) / 256, 256, 0, stream>>>(
            xz, cx_w + (size_t)d * Iq * 3, cz_w + (size_t)d * Iq * 3,
            xbc, yz, d);
        // 2c) low-rank projections + delta/bt/ct
        proj_delta_kernel<<<Mq, 256, 0, stream>>>(
            xbc, xp_w + (size_t)d * (DTRq + 2 * Sq) * Iq,
            dt_w + (size_t)d * Iq * DTRq, dt_b + (size_t)d * Iq,
            delta, btb, ctb);
        // 2d) sequential selective scan (backward iterates l in reverse)
        scan_kernel<<<Bq, 256, 0, stream>>>(
            delta, xbc, btb, ctb, Alog + (size_t)d * Iq * Sq,
            Dp + (size_t)d * Iq, yz, d);
        // 2e) out-projection: cat[:, d*512:(d+1)*512] = [y|zb] @ out_w[d].T
        pack_a_kernel<<<packA512Blks, 256, 0, stream>>>(yz, Dq, yzpack, KT512);
        gemm_wmma_packed<<<gemmBlks, 256, 0, stream>>>(
            yzpack, outwp[d], cat + (size_t)d * Dq, 3 * Dq, KT512, tilesN);
    }

    // 3) mix: mixed = cat @ mix_w.T   (K = 1536)
    pack_a_kernel<<<packA1536Blks, 256, 0, stream>>>(cat, 3 * Dq, catpack, KT1536);
    gemm_wmma_packed<<<gemmBlks, 256, 0, stream>>>(
        catpack, mixwp, mixed, Dq, KT1536, tilesN);

    // 4) output LayerNorm
    ln_out_kernel<<<Mq, 256, 0, stream>>>(mixed, ln_out_w, ln_out_b, outp);
}
